// MultiHeadSelfAttention_34961033790036
// MI455X (gfx1250) — compile-verified, matrix operand order checked by reading, not run on hardware
//
#include <hip/hip_runtime.h>

typedef _Float16 half_t;
typedef __attribute__((ext_vector_type(16))) _Float16 v16h;
typedef __attribute__((ext_vector_type(8)))  float    v8f;
typedef unsigned int u32;
typedef unsigned long long u64;
typedef unsigned int u32x4 __attribute__((ext_vector_type(4)));
typedef int          i32x8 __attribute__((ext_vector_type(8)));
typedef int          i32x4 __attribute__((ext_vector_type(4)));

#define B_  4
#define N_  2048
#define C_  1024
#define H_  16
#define D_  64

union FragU { v16h v; u32 u[8]; };

__device__ __forceinline__ u32 pack2(float a, float b) {
  union { half_t h[2]; u32 u; } x;
  x.h[0] = (half_t)a; x.h[1] = (half_t)b;
  return x.u;
}

// Load a 16x32 f16 A-fragment / 32x16 B-fragment from an LDS tile stored as
// [row][K halves]. Per CDNA5 ISA 7.12.2 both layouts reduce to: row = lane%16,
// halves e=0..15 cover K = (e/8)*16 + (lane/16)*8 + (e%8)  -> two 16B chunks.
__device__ __forceinline__ v16h load_frag(const u32* lds, int row_base,
                                          int stride_u32, int kbase_u32) {
  const int lane = threadIdx.x & 31;
  const int r  = row_base + (lane & 15);
  const int g4 = (lane >> 4) << 2;
  const u32* p = lds + r * stride_u32 + kbase_u32;
  FragU f;
#pragma unroll
  for (int i = 0; i < 4; ++i) {
    f.u[i]     = p[g4 + i];
    f.u[4 + i] = p[8 + g4 + i];
  }
  return f.v;
}

__device__ __forceinline__ v8f wmma_f16(v16h a, v16h b, v8f c) {
  return __builtin_amdgcn_wmma_f32_16x16x32_f16(false, a, false, b,
                                                (short)0, c, false, false);
}

// ---------------------------------------------------------------------------
// Tensor Data Mover: async 2D f16 tile load global->LDS with row padding.
// D# per ISA ch.8: group0 = {count=1, lds_addr, global_addr(57b), type=2};
// group1 packs data_size=2B, pad_enable, pad_interval/pad_amount codes,
// tensor dims, tile dims, dim0 stride. Groups 2/3 zero (2D tensor).
// pad codes: interval c -> pad after 2^(c+1) DWORDs; amount c -> (c+1) DWORDs.
// This toolchain's builtin takes 6 args (g0, g1, g2, g3, extra, cpol).
// ---------------------------------------------------------------------------
__device__ __forceinline__ void tdm_load_2d(u32 lds_off, const void* gptr,
                                            u32 tile_d0, u32 tile_d1,
                                            u64 stride0_elems,
                                            u32 tensor_d0, u32 tensor_d1,
                                            u32 pad_int_code, u32 pad_amt_code) {
  u64 ga = (u64)(size_t)gptr;
  u32x4 g0 = { 1u,                                   // count=1 (valid user D#)
               lds_off,
               (u32)ga,
               (u32)(ga >> 32) | (2u << 30) };       // type=2 ("image")
  i32x8 g1;
  g1[0] = (int)((1u << 16)                           // data_size: 2 bytes
              | (1u << 20)                           // pad_enable
              | (pad_int_code << 22)
              | (pad_amt_code << 25));
  g1[1] = (int)((tensor_d0 & 0xFFFFu) << 16);
  g1[2] = (int)((tensor_d0 >> 16) | ((tensor_d1 & 0xFFFFu) << 16));
  g1[3] = (int)((tensor_d1 >> 16) | (tile_d0 << 16));
  g1[4] = (int)(tile_d1 & 0xFFFFu);                  // tile_dim1 (tile_dim2=0)
  g1[5] = (int)(u32)stride0_elems;
  g1[6] = (int)(u32)((stride0_elems >> 32) & 0xFFFFu);
  g1[7] = 0;
  i32x4 gz4 = { 0, 0, 0, 0 };
  i32x8 gz8 = { 0, 0, 0, 0, 0, 0, 0, 0 };
  __builtin_amdgcn_tensor_load_to_lds(g0, g1, gz4, gz4, gz8, 0);
}

__device__ __forceinline__ u32 lds_off(const void* p) {
  return (u32)(size_t)p;   // generic LDS pointer: offset = addr[31:0] (ISA 10.2)
}

// ---------------------------------------------------------------------------
// Pre-pass 1: f32 -> f16 elementwise (for x).
// ---------------------------------------------------------------------------
__global__ __launch_bounds__(256)
void cvt_f16(const float* __restrict__ in, half_t* __restrict__ out) {
  int i = blockIdx.x * 256 + threadIdx.x;
  float4 d = ((const float4*)in)[i];
  uint2 o;
  o.x = pack2(d.x, d.y);
  o.y = pack2(d.z, d.w);
  ((uint2*)out)[i] = o;
}

// ---------------------------------------------------------------------------
// Pre-pass 2: W f32 [K][Nc] -> Wt f16 [Nc][K]  (64x64 tiles via LDS).
// ---------------------------------------------------------------------------
__global__ __launch_bounds__(256)
void transpose_cvt(const float* __restrict__ W, half_t* __restrict__ Wt,
                   int K, int Nc) {
  __shared__ __align__(16) half_t Lh[64 * 72];
  const int tid = threadIdx.x;
  const int k0 = blockIdx.x * 64, n0 = blockIdx.y * 64;
#pragma unroll
  for (int i = 0; i < 4; ++i) {
    int idx = tid + i * 256;                 // 1024 float4
    int kr  = idx >> 4;                      // 16 float4 per k-row
    int c   = idx & 15;
    float4 d = *((const float4*)(W + (size_t)(k0 + kr) * Nc + n0) + c);
    int nb = c * 4;
    Lh[(nb + 0) * 72 + kr] = (half_t)d.x;
    Lh[(nb + 1) * 72 + kr] = (half_t)d.y;
    Lh[(nb + 2) * 72 + kr] = (half_t)d.z;
    Lh[(nb + 3) * 72 + kr] = (half_t)d.w;
  }
  __syncthreads();
#pragma unroll
  for (int i = 0; i < 2; ++i) {
    int idx = tid + i * 256;                 // 512 uint4
    int nr  = idx >> 3;                      // 8 uint4 per n-row
    int c   = idx & 7;
    uint4 d = *(const uint4*)&Lh[nr * 72 + c * 8];
    *((uint4*)(Wt + (size_t)(n0 + nr) * K + k0) + c) = d;
  }
}

// ---------------------------------------------------------------------------
// GEMM: out[M,Nc] = A[M,K](f16) * Bt[Nc,K](f16, pre-transposed) + bias.
// OUT_MODE 0: f16 -> [B,H,N,D]; 1: f32 row-major; 2: f16 -> [B,H,D,N] (V^T).
// 256 thr = 8 waves (2M x 4N); tile 128x128, K-step 32.
// TDM double-buffered staging: wave0 issues next A/B tile pair, waits
// tensorcnt<=2 for current pair, barrier, 8 WMMAs, barrier.
// ---------------------------------------------------------------------------
template <int OUT_MODE>
__global__ __launch_bounds__(256)
void gemm_tdm(const half_t* __restrict__ A,
              const half_t* __restrict__ Bt,
              const float* __restrict__ bias,
              void* __restrict__ out,
              int M, int K, int Nc) {
  __shared__ __align__(16) u32 As[2][128 * 20];   // [row][16 u32], TDM-padded to 20
  __shared__ __align__(16) u32 Bs[2][128 * 20];

  const int tid  = threadIdx.x;
  const int lane = tid & 31;
  const int wid  = tid >> 5;
  const int wm   = wid >> 2;
  const int wn   = wid & 3;
  const int m0   = blockIdx.x * 128;
  const int n0   = blockIdx.y * 128;

  v8f acc[4][2];
#pragma unroll
  for (int i = 0; i < 4; ++i)
#pragma unroll
    for (int j = 0; j < 2; ++j) acc[i][j] = (v8f)0.0f;

  const half_t* Abase = A  + (size_t)m0 * K;
  const half_t* Bbase = Bt + (size_t)n0 * K;

  if (tid < 32) {   // wave 0 drives the TDM pipeline
    tdm_load_2d(lds_off(&As[0][0]), Abase, 32, 128, (u64)K, (u32)K, (u32)M, 3, 3);
    tdm_load_2d(lds_off(&Bs[0][0]), Bbase, 32, 128, (u64)K, (u32)K, (u32)Nc, 3, 3);
  }

  const int T = K / 32;
  for (int t = 0; t < T; ++t) {
    const int cur = t & 1, nxt = cur ^ 1;
    if (tid < 32) {
      if (t + 1 < T) {
        const int k1 = (t + 1) * 32;
        tdm_load_2d(lds_off(&As[nxt][0]), Abase + k1, 32, 128, (u64)K, (u32)K, (u32)M, 3, 3);
        tdm_load_2d(lds_off(&Bs[nxt][0]), Bbase + k1, 32, 128, (u64)K, (u32)K, (u32)Nc, 3, 3);
        __builtin_amdgcn_s_wait_tensorcnt((short)2);   // current pair landed
      } else {
        __builtin_amdgcn_s_wait_tensorcnt((short)0);
      }
    }
    __syncthreads();

    v16h a[4];
#pragma unroll
    for (int mb = 0; mb < 4; ++mb)
      a[mb] = load_frag(As[cur], wm * 64 + mb * 16, 20, 0);
    v16h b[2];
#pragma unroll
    for (int nb = 0; nb < 2; ++nb)
      b[nb] = load_frag(Bs[cur], wn * 32 + nb * 16, 20, 0);
#pragma unroll
    for (int mb = 0; mb < 4; ++mb)
#pragma unroll
      for (int nb = 0; nb < 2; ++nb)
        acc[mb][nb] = wmma_f16(a[mb], b[nb], acc[mb][nb]);
    __syncthreads();   // reads done before next TDM overwrites this buffer
  }

  // ---- epilogue: bias + store ----
  const int hi = lane >> 4;
  const int lo = lane & 15;
#pragma unroll
  for (int mb = 0; mb < 4; ++mb) {
#pragma unroll
    for (int nb = 0; nb < 2; ++nb) {
      int gn = n0 + wn * 32 + nb * 16 + lo;
      float bv = bias[gn];
#pragma unroll
      for (int j = 0; j < 8; ++j) {
        int gm = m0 + wm * 64 + mb * 16 + j + 8 * hi;
        float v = acc[mb][nb][j] + bv;
        if (OUT_MODE == 0) {        // f16 [B,H,N,D]
          int bb = gm >> 11, nn = gm & (N_ - 1);
          int hh = gn >> 6,  dd = gn & (D_ - 1);
          ((half_t*)out)[(((size_t)(bb * H_ + hh)) * N_ + nn) * D_ + dd] = (half_t)v;
        } else if (OUT_MODE == 2) { // f16 [B,H,D,N] (pre-transposed V)
          int bb = gm >> 11, nn = gm & (N_ - 1);
          int hh = gn >> 6,  dd = gn & (D_ - 1);
          ((half_t*)out)[(((size_t)(bb * H_ + hh)) * D_ + dd) * N_ + nn] = (half_t)v;
        } else {                    // f32 row-major
          ((float*)out)[(size_t)gm * Nc + gn] = v;
        }
      }
    }
  }
}

// ---------------------------------------------------------------------------
// Flash attention: block per (b*H+h, 64-query tile); 128 thr = 4 waves, wave w
// owns q rows [16w,16w+16). Q/K/V^T tiles all staged by TDM (K,V^T double-
// buffered); online softmax in f32; P round-trips LDS as f16 A-fragments.
// V is pre-transposed in global memory: Vg = [B,H,D,N].
// ---------------------------------------------------------------------------
__global__ __launch_bounds__(128)
void flash_attn(const half_t* __restrict__ Q,
                const half_t* __restrict__ Km,
                const half_t* __restrict__ Vg,
                half_t* __restrict__ ctx) {
  __shared__ __align__(16) u32 Qs[64 * 36];        // [q][d]
  __shared__ __align__(16) u32 Ks[2][64 * 36];     // [key][d]
  __shared__ __align__(16) u32 Vt[2][64 * 36];     // [d][key]
  __shared__ __align__(16) u32 Ps[64 * 36];        // [q][key]

  const int tid  = threadIdx.x;
  const int lane = tid & 31;
  const int w    = tid >> 5;
  const int hi   = lane >> 4;
  const int lo   = lane & 15;

  const int bh = blockIdx.y;
  const int q0 = blockIdx.x * 64;

  const half_t* Qh = Q  + (size_t)bh * N_ * D_;
  const half_t* Kh = Km + (size_t)bh * N_ * D_;
  const half_t* Vh = Vg + (size_t)bh * D_ * N_;    // [D][N]

  if (tid < 32) {
    tdm_load_2d(lds_off(&Qs[0]),    Qh + (size_t)q0 * D_, 64, 64, (u64)D_, D_, N_, 4, 3);
    tdm_load_2d(lds_off(&Ks[0][0]), Kh,                   64, 64, (u64)D_, D_, N_, 4, 3);
    tdm_load_2d(lds_off(&Vt[0][0]), Vh,                   64, 64, (u64)N_, N_, D_, 4, 3);
  }

  v8f o[4];
#pragma unroll
  for (int nd = 0; nd < 4; ++nd) o[nd] = (v8f)0.0f;
  float mrow[8], lrow[8];
#pragma unroll
  for (int j = 0; j < 8; ++j) { mrow[j] = -__builtin_inff(); lrow[j] = 0.0f; }

  const float scale = 0.125f;                      // 1/sqrt(64)
  const int T = N_ / 64;

  for (int t = 0; t < T; ++t) {
    const int cur = t & 1, nxt = cur ^ 1;
    if (tid < 32) {
      if (t + 1 < T) {
        const int k1 = (t + 1) * 64;
        tdm_load_2d(lds_off(&Ks[nxt][0]), Kh + (size_t)k1 * D_, 64, 64, (u64)D_, D_, N_, 4, 3);
        tdm_load_2d(lds_off(&Vt[nxt][0]), Vh + k1,              64, 64, (u64)N_, N_, D_, 4, 3);
        __builtin_amdgcn_s_wait_tensorcnt((short)2);
      } else {
        __builtin_amdgcn_s_wait_tensorcnt((short)0);
      }
    }
    __syncthreads();

    // ---- S = Q * K^T ----
    v16h a0 = load_frag(Qs, w * 16, 36, 0);
    v16h a1 = load_frag(Qs, w * 16, 36, 16);
    v8f s[4];
#pragma unroll
    for (int nb = 0; nb < 4; ++nb) {
      v16h b0 = load_frag(Ks[cur], nb * 16, 36, 0);
      v16h b1 = load_frag(Ks[cur], nb * 16, 36, 16);
      v8f tacc = (v8f)0.0f;
      tacc = wmma_f16(a0, b0, tacc);
      tacc = wmma_f16(a1, b1, tacc);
      s[nb] = tacc;
    }

    // ---- online softmax: row = 16w + j + 8*hi ----
    half_t* Ph = (half_t*)Ps;
#pragma unroll
    for (int j = 0; j < 8; ++j) {
      float rmax = -__builtin_inff();
#pragma unroll
      for (int nb = 0; nb < 4; ++nb) {
        float v = s[nb][j] * scale;
        s[nb][j] = v;
        rmax = fmaxf(rmax, v);
      }
#pragma unroll
      for (int off = 8; off >= 1; off >>= 1)
        rmax = fmaxf(rmax, __shfl_xor(rmax, off, 32));
      float mnew  = fmaxf(mrow[j], rmax);
      float alpha = __expf(mrow[j] - mnew);
      float rsum  = 0.0f;
#pragma unroll
      for (int nb = 0; nb < 4; ++nb) {
        float p = __expf(s[nb][j] - mnew);
        rsum += p;
        Ph[(w * 16 + j + 8 * hi) * 72 + nb * 16 + lo] = (half_t)p;
      }
#pragma unroll
      for (int off = 8; off >= 1; off >>= 1)
        rsum += __shfl_xor(rsum, off, 32);
      lrow[j] = lrow[j] * alpha + rsum;
      mrow[j] = mnew;
#pragma unroll
      for (int nd = 0; nd < 4; ++nd)
        o[nd][j] *= alpha;
    }
    __syncthreads();

    // ---- O += P * V ----
    v16h p0 = load_frag(Ps, w * 16, 36, 0);
    v16h p1 = load_frag(Ps, w * 16, 36, 16);
#pragma unroll
    for (int nd = 0; nd < 4; ++nd) {
      v16h b0 = load_frag(Vt[cur], nd * 16, 36, 0);
      v16h b1 = load_frag(Vt[cur], nd * 16, 36, 16);
      o[nd] = wmma_f16(p0, b0, o[nd]);
      o[nd] = wmma_f16(p1, b1, o[nd]);
    }
    __syncthreads();
  }

  // ---- epilogue: O /= l, store ctx f16 [B,N,C] ----
  const int bb = bh >> 4;
  const int hh = bh & 15;
#pragma unroll
  for (int nd = 0; nd < 4; ++nd) {
#pragma unroll
    for (int j = 0; j < 8; ++j) {
      int q = q0 + w * 16 + j + 8 * hi;
      int d = nd * 16 + lo;
      float v = o[nd][j] / lrow[j];
      ctx[((size_t)(bb * N_ + q)) * C_ + hh * D_ + d] = (half_t)v;
    }
  }
}

// ---------------------------------------------------------------------------
extern "C" void kernel_launch(void* const* d_in, const int* in_sizes, int n_in,
                              void* d_out, int out_size, void* d_ws, size_t ws_size,
                              hipStream_t stream) {
  (void)in_sizes; (void)n_in; (void)out_size; (void)ws_size;
  const float* x  = (const float*)d_in[0];
  const float* Wq = (const float*)d_in[1];
  const float* bq = (const float*)d_in[2];
  const float* Wk = (const float*)d_in[3];
  const float* bk = (const float*)d_in[4];
  const float* Wv = (const float*)d_in[5];
  const float* bv = (const float*)d_in[6];
  const float* Wo = (const float*)d_in[7];
  const float* bo = (const float*)d_in[8];
  float* out = (float*)d_out;

  const size_t EB = (size_t)B_ * N_ * C_;       // 8,388,608 (big f16 buffers)
  const size_t EW = (size_t)C_ * C_;            // 1,048,576 (weight buffers)
  half_t* xh  = (half_t*)d_ws;                  // f16 [B*N, C]
  half_t* Wtq = xh  + EB;                       // f16 [C, C] transposed
  half_t* Wtk = Wtq + EW;
  half_t* Wtv = Wtk + EW;
  half_t* Wto = Wtv + EW;
  half_t* Qf  = Wto + EW;                       // f16 [B,H,N,D]
  half_t* Kf  = Qf  + EB;                       // f16 [B,H,N,D]
  half_t* Vtg = Kf  + EB;                       // f16 [B,H,D,N]  (pre-transposed)
  half_t* ctx = Vtg + EB;                       // f16 [B,N,C]

  const int M = B_ * N_;                        // 8192

  cvt_f16<<<(int)(EB / 4 / 256), 256, 0, stream>>>(x, xh);
  dim3 gT(C_ / 64, C_ / 64);
  transpose_cvt<<<gT, 256, 0, stream>>>(Wq, Wtq, C_, C_);
  transpose_cvt<<<gT, 256, 0, stream>>>(Wk, Wtk, C_, C_);
  transpose_cvt<<<gT, 256, 0, stream>>>(Wv, Wtv, C_, C_);
  transpose_cvt<<<gT, 256, 0, stream>>>(Wo, Wto, C_, C_);

  dim3 gGemm(M / 128, C_ / 128);
  gemm_tdm<0><<<gGemm, 256, 0, stream>>>(xh, Wtq, bq, Qf,  M, C_, C_);
  gemm_tdm<0><<<gGemm, 256, 0, stream>>>(xh, Wtk, bk, Kf,  M, C_, C_);
  gemm_tdm<2><<<gGemm, 256, 0, stream>>>(xh, Wtv, bv, Vtg, M, C_, C_);

  dim3 gAttn(N_ / 64, B_ * H_);
  flash_attn<<<gAttn, 128, 0, stream>>>(Qf, Kf, Vtg, ctx);

  gemm_tdm<1><<<gGemm, 256, 0, stream>>>(ctx, Wto, bo, out, M, C_, C_);
}
